// Texture_LMSC_90787018703566
// MI455X (gfx1250) — matrix operand order repaired and checked by reading, not compiled, not run on hardware
//
#include <hip/hip_runtime.h>

typedef __attribute__((ext_vector_type(16))) _Float16 v16h;
typedef __attribute__((ext_vector_type(8)))  _Float16 v8h;
typedef __attribute__((ext_vector_type(8)))  float    v8f;

#define B_DIM   8
#define T_DIM   64
#define IN_DIM  6
#define G_DIM   256
#define H_DIM   32
#define W_DIM   128
#define OUT_DIM 3
#define RB      64              // rows per block (2048/64 = 32 blocks)

// fp16 transposed-weight workspace offsets (in halfs).
// Layout of each matrix: [Npad][Kpad], Kpad multiple of 32 (WMMA K-step).
#define OFF_WA0 0        // qbA_W0^T  [128][64]  (SD=38 padded to 64)
#define OFF_WB0 8192     // qbB_W0^T  [128][64]
#define OFF_WA1 16384    // qbA_Ws^T  3 x [128][128]
#define OFF_WB1 65536    // qbB_Ws^T  3 x [128][128]
#define OFF_AL  114688   // alpha_W^T [32][128]
#define OFF_BE  118784   // beta_W^T  [32][128]
#define OFF_O0  122880   // out_W0^T  [128][32]
#define OFF_O1  126976   // out_W1^T  [128][128]
#define OFF_O2  143360   // out_W2^T  [16][128]  (N=3 padded to 16)

// LDS bias table offsets (floats)
#define LB_A0 0     // 128
#define LB_B0 128   // 128
#define LB_AS 256   // 3*128
#define LB_BS 640   // 3*128
#define LB_AL 1024  // 32
#define LB_BE 1056  // 32
#define LB_O0 1088  // 128
#define LB_O1 1216  // 128
#define LB_O2 1344  // 16 (padded, cols>=3 are zero)
#define LB_TOTAL 1360

// ---------------------------------------------------------------------------
// fp32 [K][N] -> fp16 [Npad][Kpad] transpose + zero-pad (B-operand layout:
// lane n reads 16 contiguous K-halfs -> one 32B load per WMMA).
__global__ void prep_w(const float* __restrict__ src, _Float16* __restrict__ dst,
                       int K, int N, int Kpad, int Npad) {
  const int i = blockIdx.x * blockDim.x + threadIdx.x;
  const int total = Kpad * Npad;
  if (i >= total) return;
  const int n = i / Kpad, k = i % Kpad;
  const float v = (k < K && n < N) ? src[(size_t)k * N + n] : 0.0f;
  dst[(size_t)n * Kpad + k] = (_Float16)v;
}

__device__ __forceinline__ float fast_tanh(float x) {
  x = fminf(fmaxf(x, -15.0f), 15.0f);
  const float e = __expf(2.0f * x);
  return (e - 1.0f) / (e + 1.0f);
}

// ---------------------------------------------------------------------------
// A-operand fetch from LDS (row-major fp16, leading dim lda), 16-row M-tile.
//   A layout used here (contiguous K, as B): lanes 0-15 row m, K 0..15; lanes 16-31 same rows,
//   K 16..31  -> two 16B ds_loads per K-step.
template<int KSTEPS>
__device__ __forceinline__ void load_a_ops(const _Float16* asrc, int lda,
                                           int rowbase, int lane, v16h A[KSTEPS]) {
  const int m  = lane & 15;
  const int hi = (lane >> 4) & 1;
#pragma unroll
  for (int kt = 0; kt < KSTEPS; ++kt) {
    const _Float16* p = asrc + (rowbase + m) * lda + kt * 32 + hi * 16;
    union { v16h v; struct { v8h lo, hi; } s; } u;
    u.s.lo = *(const v8h*)(p);
    u.s.hi = *(const v8h*)(p + 8);
    A[kt] = u.v;
  }
}

// One 16x16 output tile: acc = A @ Wt[:,col-tile]  (zero C -> first WMMA uses
// the inline-0 SRC2; bias is added in the caller's epilogue).
// B layout: lanes 0-15 hold K 0..15 of col n, lanes 16-31 K 16..31 -> 32B load.
template<int KSTEPS>
__device__ __forceinline__ v8f wmma_col(const v16h A[KSTEPS],
                                        const _Float16* __restrict__ wt,
                                        int col, int hi) {
  v8f c = { 0.f, 0.f, 0.f, 0.f, 0.f, 0.f, 0.f, 0.f };
#pragma unroll
  for (int kt = 0; kt < KSTEPS; ++kt) {
    const v16h b = *(const v16h*)(wt + (size_t)col * (KSTEPS * 32) + kt * 32 + hi * 16);
    c = __builtin_amdgcn_wmma_f32_16x16x32_f16(false, A[kt], false, b,
                                               (short)0, c, false, false);
  }
  return c;
}

// Scatter one f32 16x16 C/D tile into LDS as fp16 (row-major, leading dim ld).
// C/D layout: VGPR e, lanes 0-15 -> row e, col lane; lanes 16-31 -> row 8+e.
__device__ __forceinline__ void store_tile_f16(_Float16* dst, int ld, int rowbase,
                                               int col, int lane, v8f v) {
  const int r0 = rowbase + ((lane >> 4) & 1) * 8;
#pragma unroll
  for (int i = 0; i < 8; ++i)
    dst[(r0 + i) * ld + col] = (_Float16)v[i];
}

// ---------------------------------------------------------------------------
// Persistent fused RNN kernel: 32 blocks x 64 rows; 8 waves per block.
// Wave w: rows (w&3)*16..+16, column half (w>>2): col tiles (w>>2)*4..+4.
__global__ __launch_bounds__(256)
void lmsc_main(const float* __restrict__ x,
               const float* __restrict__ init_ori,
               const _Float16* __restrict__ ws,
               const float* __restrict__ bA0, const float* __restrict__ bAs,
               const float* __restrict__ bB0, const float* __restrict__ bBs,
               const float* __restrict__ alb, const float* __restrict__ beb,
               const float* __restrict__ ob0, const float* __restrict__ ob1,
               const float* __restrict__ ob2,
               float* __restrict__ out)
{
  extern __shared__ __align__(32) _Float16 smem[];
  _Float16* zA   = smem;                          // [RB][128] fp16
  _Float16* zB   = smem + RB * 128;               // [RB][128] fp16 (also z0 [RB][64])
  float*    hbuf = (float*)(smem + 2 * RB * 128); // [RB][32] f32 state
  float*    bls  = hbuf + RB * 32;                // bias table (LB_TOTAL floats)

  const int tid     = threadIdx.x;
  const int lane    = tid & 31;
  const int wave    = tid >> 5;
  const int rowbase = (wave & 3) * 16;   // 4 row tiles
  const int cg      = wave >> 2;         // column group: 0 or 1
  const int ntbase  = cg * 4;            // of 8 column tiles in 128-wide layers
  const int m       = lane & 15;
  const int hi      = (lane >> 4) & 1;
  const int r0      = blockIdx.x * RB;
  const int b       = r0 / G_DIM;        // 64 | 256: one batch index per block
  const int g0      = r0 % G_DIM;

  // ---- preload biases into LDS (out_b2 zero-padded to 16 cols)
  for (int i = tid; i < LB_TOTAL; i += 256) {
    float v = 0.0f;
    if (i < LB_B0)                       v = bA0[i - LB_A0];
    else if (i < LB_AS)                  v = bB0[i - LB_B0];
    else if (i < LB_BS)                  v = bAs[i - LB_AS];
    else if (i < LB_AL)                  v = bBs[i - LB_BS];
    else if (i < LB_BE)                  v = alb[i - LB_AL];
    else if (i < LB_O0)                  v = beb[i - LB_BE];
    else if (i < LB_O1)                  v = ob0[i - LB_O0];
    else if (i < LB_O2)                  v = ob1[i - LB_O1];
    else if (i - LB_O2 < OUT_DIM)        v = ob2[i - LB_O2];
    bls[i] = v;
  }

  // ---- h0 = 0, first 3 channels = initial orientations
  for (int idx = tid; idx < RB * 32; idx += 256) {
    const int r = idx >> 5, c = idx & 31;
    float v = 0.0f;
    if (c < 3) v = init_ori[(size_t)(b * G_DIM + g0 + r) * 3 + c];
    hbuf[idx] = v;
  }
  __syncthreads();

#pragma unroll 1
  for (int t = 0; t < T_DIM; ++t) {
    // strain norm + direction (uniform across the block)
    float xv[IN_DIM], ss = 0.0f;
#pragma unroll
    for (int i = 0; i < IN_DIM; ++i) {
      xv[i] = x[(size_t)(b * T_DIM + t) * IN_DIM + i];
      ss += xv[i] * xv[i];
    }
    const float sn  = sqrtf(ss);
    const float inv = 1.0f / (sn + 1e-15f);

    // z0 = [xn | h | 0 pad] fp16, [RB][64] in the zB region
    _Float16* z0 = zB;
#pragma unroll 1
    for (int idx = tid; idx < RB * 64; idx += 256) {
      const int r = idx >> 6, c = idx & 63;
      float v = 0.0f;
      if (c < IN_DIM)              v = xv[c] * inv;
      else if (c < IN_DIM + H_DIM) v = hbuf[r * 32 + (c - IN_DIM)];
      z0[idx] = (_Float16)v;
    }
    __syncthreads();

    // ---- quadratic block layer 0: [*,64]x[64,128]; tanh(a)*tanh(c)
    {
      v16h A[2];
      load_a_ops<2>(z0, 64, rowbase, lane, A);
#pragma unroll 1
      for (int nt = 0; nt < 4; ++nt) {
        const int col = (ntbase + nt) * 16 + m;
        v8f a = wmma_col<2>(A, ws + OFF_WA0, col, hi);
        v8f c = wmma_col<2>(A, ws + OFF_WB0, col, hi);
        const float bva = bls[LB_A0 + col], bvc = bls[LB_B0 + col];
        v8f z;
#pragma unroll
        for (int e = 0; e < 8; ++e)
          z[e] = fast_tanh(a[e] + bva) * fast_tanh(c[e] + bvc);
        store_tile_f16(zA, 128, rowbase, col, lane, z);
        __builtin_amdgcn_sched_barrier(0);
      }
    }
    __syncthreads();

    // ---- quadratic layers 1..3: [*,128]x[128,128], ping-pong zA<->zB
    _Float16* zsrc = zA;
    _Float16* zdst = zB;
#pragma unroll 1
    for (int j = 0; j < 3; ++j) {
      v16h A[4];
      load_a_ops<4>(zsrc, 128, rowbase, lane, A);
      const _Float16* wtA = ws + OFF_WA1 + j * 16384;
      const _Float16* wtB = ws + OFF_WB1 + j * 16384;
      const bool mid = (j < 2);
#pragma unroll 1
      for (int nt = 0; nt < 4; ++nt) {
        const int col = (ntbase + nt) * 16 + m;
        v8f a = wmma_col<4>(A, wtA, col, hi);
        v8f c = wmma_col<4>(A, wtB, col, hi);
        const float bva = bls[LB_AS + j * 128 + col];
        const float bvc = bls[LB_BS + j * 128 + col];
        v8f z;
#pragma unroll
        for (int e = 0; e < 8; ++e) {
          float av = a[e] + bva, cv = c[e] + bvc;
          if (mid) { av = fast_tanh(av); cv = fast_tanh(cv); }
          float p = av * cv;
          if (!mid) p = fast_tanh(p);       // fold q = tanh(z) into last layer
          z[e] = p;
        }
        store_tile_f16(zdst, 128, rowbase, col, lane, z);
        __builtin_amdgcn_sched_barrier(0);
      }
      __syncthreads();
      _Float16* tmp = zsrc; zsrc = zdst; zdst = tmp;
    }
    const _Float16* qbuf = zsrc;            // q in zB

    // ---- alpha / beta heads + gated state update; wave group cg owns
    //      h-columns cg*16..cg*16+15 (disjoint between wave pairs)
    {
      v16h A[4];
      load_a_ops<4>(qbuf, 128, rowbase, lane, A);
      const int col = cg * 16 + m;
      v8f ua = wmma_col<4>(A, ws + OFF_AL, col, hi);
      v8f ub = wmma_col<4>(A, ws + OFF_BE, col, hi);
      const float bva = bls[LB_AL + col], bvc = bls[LB_BE + col];
      const int rbase2 = rowbase + hi * 8;
#pragma unroll
      for (int e = 0; e < 8; ++e) {
        const int r       = rbase2 + e;
        const float alpha = __expf(ua[e] + bva);
        const float beta  = fast_tanh(ub[e] + bvc);
        const float hold  = hbuf[r * 32 + col];
        const float dec   = __expf(-alpha * sn);
        const float hn    = dec * (hold - beta) + beta;
        hbuf[r * 32 + col] = hn;
        zA[r * 32 + col]   = (_Float16)hn;  // fp16 copy, [RB][32] A layout
      }
    }
    __syncthreads();

    // ---- output MLP layer 0: [*,32]x[32,128], tanh
    {
      v16h A[1];
      load_a_ops<1>(zA, 32, rowbase, lane, A);
#pragma unroll 1
      for (int nt = 0; nt < 4; ++nt) {
        const int col = (ntbase + nt) * 16 + m;
        v8f a = wmma_col<1>(A, ws + OFF_O0, col, hi);
        const float bv = bls[LB_O0 + col];
        v8f z;
#pragma unroll
        for (int e = 0; e < 8; ++e) z[e] = fast_tanh(a[e] + bv);
        store_tile_f16(zB, 128, rowbase, col, lane, z);
        __builtin_amdgcn_sched_barrier(0);
      }
    }
    __syncthreads();

    // ---- output MLP layer 1: [*,128]x[128,128], tanh
    {
      v16h A[4];
      load_a_ops<4>(zB, 128, rowbase, lane, A);
#pragma unroll 1
      for (int nt = 0; nt < 4; ++nt) {
        const int col = (ntbase + nt) * 16 + m;
        v8f a = wmma_col<4>(A, ws + OFF_O1, col, hi);
        const float bv = bls[LB_O1 + col];
        v8f z;
#pragma unroll
        for (int e = 0; e < 8; ++e) z[e] = fast_tanh(a[e] + bv);
        store_tile_f16(zA, 128, rowbase, col, lane, z);
        __builtin_amdgcn_sched_barrier(0);
      }
    }
    __syncthreads();   // o2 columns come from both wave groups

    // ---- output MLP layer 2: [*,128]x[128,3] (N padded to 16); cg==0 only
    if (cg == 0) {     // wave-uniform: EXEC stays all-ones for the WMMA
      v16h A[4];
      load_a_ops<4>(zA, 128, rowbase, lane, A);
      v8f o = wmma_col<4>(A, ws + OFF_O2, m, hi);
      const float bv = bls[LB_O2 + m];
      if (m < OUT_DIM) {
        const int rbase2 = rowbase + hi * 8;
#pragma unroll
        for (int e = 0; e < 8; ++e) {
          const int g = g0 + rbase2 + e;
          out[(((size_t)(b * G_DIM + g)) * T_DIM + t) * OUT_DIM + m] = o[e] + bv;
        }
      }
    }
    __syncthreads();   // protect zB/hbuf before next step's rebuild
  }
}

// ---------------------------------------------------------------------------
extern "C" void kernel_launch(void* const* d_in, const int* in_sizes, int n_in,
                              void* d_out, int out_size, void* d_ws, size_t ws_size,
                              hipStream_t stream) {
  const float* x        = (const float*)d_in[0];
  const float* init_ori = (const float*)d_in[1];
  const float* qbA_W0   = (const float*)d_in[2];
  const float* qbA_b0   = (const float*)d_in[3];
  const float* qbA_Ws   = (const float*)d_in[4];
  const float* qbA_bs   = (const float*)d_in[5];
  const float* qbB_W0   = (const float*)d_in[6];
  const float* qbB_b0   = (const float*)d_in[7];
  const float* qbB_Ws   = (const float*)d_in[8];
  const float* qbB_bs   = (const float*)d_in[9];
  const float* alpha_W  = (const float*)d_in[10];
  const float* alpha_b  = (const float*)d_in[11];
  const float* beta_W   = (const float*)d_in[12];
  const float* beta_b   = (const float*)d_in[13];
  const float* out_W0   = (const float*)d_in[14];
  const float* out_b0   = (const float*)d_in[15];
  const float* out_W1   = (const float*)d_in[16];
  const float* out_b1   = (const float*)d_in[17];
  const float* out_W2   = (const float*)d_in[18];
  const float* out_b2   = (const float*)d_in[19];

  _Float16* ws = (_Float16*)d_ws;

  auto prep = [&](const float* src, size_t off, int K, int N, int Kpad, int Npad) {
    const int total = Kpad * Npad;
    prep_w<<<(total + 255) / 256, 256, 0, stream>>>(src, ws + off, K, N, Kpad, Npad);
  };
  prep(qbA_W0, OFF_WA0, 38, 128, 64, 128);
  prep(qbB_W0, OFF_WB0, 38, 128, 64, 128);
  for (int j = 0; j < 3; ++j) {
    prep(qbA_Ws + (size_t)j * 128 * 128, OFF_WA1 + (size_t)j * 16384, 128, 128, 128, 128);
    prep(qbB_Ws + (size_t)j * 128 * 128, OFF_WB1 + (size_t)j * 16384, 128, 128, 128, 128);
  }
  prep(alpha_W, OFF_AL, 128, 32, 128, 32);
  prep(beta_W,  OFF_BE, 128, 32, 128, 32);
  prep(out_W0,  OFF_O0, 32, 128, 32, 128);
  prep(out_W1,  OFF_O1, 128, 128, 128, 128);
  prep(out_W2,  OFF_O2, 128, 3, 128, 16);

  const size_t lds_bytes = 2 * RB * 128 * sizeof(_Float16)
                         + RB * 32 * sizeof(float)
                         + LB_TOTAL * sizeof(float);
  lmsc_main<<<(B_DIM * G_DIM) / RB, 256, lds_bytes, stream>>>(
      x, init_ori, ws,
      qbA_b0, qbA_bs, qbB_b0, qbB_bs,
      alpha_b, beta_b, out_b0, out_b1, out_b2,
      (float*)d_out);
}
